// DIEN_88201448391481
// MI455X (gfx1250) — matrix operand order repaired and checked by reading, not compiled, not run on hardware
//
#include <hip/hip_runtime.h>

typedef __attribute__((ext_vector_type(16))) __bf16 v16bf;
typedef __attribute__((ext_vector_type(8)))  __bf16 v8bf;
typedef __attribute__((ext_vector_type(8)))  float  v8f;
typedef __attribute__((ext_vector_type(4)))  float  v4f;

#define NB 2048   // batch
#define NT 100    // sequence length
#define ND 64     // embedding / hidden dim

// ---------------------------------------------------------------- helpers
__device__ __forceinline__ __bf16 to_bf16(float x) { return (__bf16)x; }  // RNE, v_cvt_pk_bf16_f32
__device__ __forceinline__ float sig_(float x) { return 1.0f / (1.0f + __expf(-x)); }

__device__ __forceinline__ v8f wmma_bf16(v16bf a, v16bf b, v8f c) {
  return __builtin_amdgcn_wmma_f32_16x16x32_bf16(false, a, false, b, (short)0, c, false, false);
}

// convert 8 contiguous fp32 (32B aligned) to v8bf
__device__ __forceinline__ v8bf cvt8(const float* p) {
  v4f a = *(const v4f*)p;
  v4f b = *(const v4f*)(p + 4);
  v8bf r;
  r[0] = (__bf16)a[0]; r[1] = (__bf16)a[1]; r[2] = (__bf16)a[2]; r[3] = (__bf16)a[3];
  r[4] = (__bf16)b[0]; r[5] = (__bf16)b[1]; r[6] = (__bf16)b[2]; r[7] = (__bf16)b[3];
  return r;
}
__device__ __forceinline__ v16bf cat16(v8bf lo, v8bf hi) {
  return __builtin_shufflevector(lo, hi, 0,1,2,3,4,5,6,7,8,9,10,11,12,13,14,15);
}

// B fragment (32x16) of row-major W[K x N]; zero-padded outside K.
// lane = column (n0+lane&15); VGPR j holds K = k0 + 2(j&3) + 16(j>>2) + 8*half (and +1).
__device__ __forceinline__ v16bf load_b_global(const float* __restrict__ W, int K, int N,
                                               int k0, int n0, int lane) {
  int n = n0 + (lane & 15), half = lane >> 4;
  v16bf b;
#pragma unroll
  for (int j = 0; j < 8; ++j) {
    int k = k0 + ((j & 3) << 1) + ((j >> 2) << 4) + (half << 3);
    b[2 * j]     = (k     < K) ? to_bf16(W[(long)k * N + n])       : to_bf16(0.f);
    b[2 * j + 1] = (k + 1 < K) ? to_bf16(W[(long)(k + 1) * N + n]) : to_bf16(0.f);
  }
  return b;
}

// B fragment from fp32 LDS tile (row-major K x N, no guards)
__device__ __forceinline__ v16bf load_b_lds_f32(const float* W, int N, int k0, int n0, int lane) {
  int n = n0 + (lane & 15), half = lane >> 4;
  v16bf b;
#pragma unroll
  for (int j = 0; j < 8; ++j) {
    int k = k0 + ((j & 3) << 1) + ((j >> 2) << 4) + (half << 3);
    b[2 * j]     = to_bf16(W[k * N + n]);
    b[2 * j + 1] = to_bf16(W[(k + 1) * N + n]);
  }
  return b;
}

// pre-swizzled fragment store/load in LDS: frag f, lane l -> 16 contiguous bf16
__device__ __forceinline__ void store_frag(__bf16* base, int f, int lane, v16bf v) {
  *(v16bf*)(base + ((f * 32 + lane) << 4)) = v;
}
__device__ __forceinline__ v16bf load_frag(const __bf16* base, int f, int lane) {
  return *(const v16bf*)(base + ((f * 32 + lane) << 4));
}

// A fragments (16x32, K0=0 and K0=32) from a contiguous 64-float row.
// element run: f[0..7] = row[h8..h8+7], f[8..15] = row[16+h8..], vectorized.
__device__ __forceinline__ void load_a_row64(v16bf& f0, v16bf& f1,
                                             const float* __restrict__ row, int lane) {
  int h8 = (lane >> 4) << 3;
  f0 = cat16(cvt8(row + h8),      cvt8(row + 16 + h8));
  f1 = cat16(cvt8(row + 32 + h8), cvt8(row + 48 + h8));
}

// A fragments from 16x64 bf16 LDS tile (row-major): four 16B ds loads
__device__ __forceinline__ void load_a_lds_bf16_64(v16bf& f0, v16bf& f1,
                                                   const __bf16* sH, int lane) {
  int m = lane & 15, h8 = (lane >> 4) << 3;
  const __bf16* r = sH + m * 64;
  f0 = cat16(*(const v8bf*)(r + h8),      *(const v8bf*)(r + 16 + h8));
  f1 = cat16(*(const v8bf*)(r + 32 + h8), *(const v8bf*)(r + 48 + h8));
}

// single A fragment from fp32 LDS tile with leading dim ld (vectorized)
__device__ __forceinline__ v16bf load_a_lds_f32(const float* s, int ld, int k0, int lane) {
  int m = lane & 15, h8 = (lane >> 4) << 3;
  const float* r = s + m * ld + k0;
  return cat16(cvt8(r + h8), cvt8(r + 16 + h8));
}

// ---------------------------------------------------------------- TDM staging
#if __has_builtin(__builtin_amdgcn_tensor_load_to_lds)
#define USE_TDM 1
typedef __attribute__((ext_vector_type(4))) unsigned tdm_v4u;
typedef __attribute__((ext_vector_type(8))) int      tdm_v8i;
typedef __attribute__((ext_vector_type(4))) int      tdm_v4i;

// DMA a full 2D fp32 tile (rows x cols, contiguous) from global into LDS.
__device__ __forceinline__ void tdm_load_2d_f32(const float* src, unsigned lds_off,
                                                unsigned rows, unsigned cols) {
  unsigned long long ga = (unsigned long long)(const void*)src;
  tdm_v4u g0;
  g0[0] = 1u;                                                  // count=1 (valid user D#)
  g0[1] = lds_off;                                             // lds_addr (bytes)
  g0[2] = (unsigned)ga;                                        // global_addr[31:0]
  g0[3] = (unsigned)((ga >> 32) & 0x01FFFFFFu) | (2u << 30);   // addr[56:32] | type=2
  tdm_v8i g1;
  g1[0] = (int)0x00020000u;                                    // data_size=2 (4 bytes)
  g1[1] = (int)((cols & 0xFFFFu) << 16);                       // tensor_dim0[15:0]
  g1[2] = (int)(((cols >> 16) & 0xFFFFu) | ((rows & 0xFFFFu) << 16)); // dim0[31:16]|dim1[15:0]
  g1[3] = (int)(((rows >> 16) & 0xFFFFu) | ((cols & 0xFFFFu) << 16)); // dim1[31:16]|tile_dim0
  g1[4] = (int)(rows & 0xFFFFu);                               // tile_dim1 (tile_dim2=0)
  g1[5] = (int)cols;                                           // tensor_dim0_stride[31:0]
  g1[6] = 0;
  g1[7] = 0;
  tdm_v4i zz4 = {0, 0, 0, 0};
  tdm_v8i zz8 = {0, 0, 0, 0, 0, 0, 0, 0};
  // 6-arg form on this toolchain: (g0, g1, g2, g3, g4, cpol)
  __builtin_amdgcn_tensor_load_to_lds(g0, g1, zz4, zz4, zz8, 0);
}
#else
#define USE_TDM 0
#endif

// ---------------------------------------------------------------- GRU / AUGRU
// One wave handles 16 batch rows for all T steps. Wx/Wh TDM-staged + swizzled to LDS frags.
template <bool AUGRU>
__global__ __launch_bounds__(32) void gru_kernel(
    const int* __restrict__ seq_idx,   // [B,T]   (!AUGRU: gather input from emb)
    const float* __restrict__ emb,     // [V,64]  (!AUGRU)
    const float* __restrict__ seq_in,  // [B,T,64] (AUGRU input = seq_h)
    const float* __restrict__ scores,  // [B,T]   (AUGRU attention)
    const int* __restrict__ hist_len,  // [B]
    const float* __restrict__ Wx, const float* __restrict__ Wh,
    const float* __restrict__ bias,    // [192]
    float* __restrict__ seq_h_out,     // [B,T,64] (!AUGRU)
    float* __restrict__ final_out)     // [B,64]   (AUGRU)
{
  __shared__ __attribute__((aligned(32))) __bf16 sFX[24 * 32 * 16];
  __shared__ __attribute__((aligned(32))) __bf16 sFH[24 * 32 * 16];
  __shared__ __attribute__((aligned(32))) __bf16 sH[16 * 64];
#if USE_TDM
  __shared__ __attribute__((aligned(16))) float sStage[64 * 192];
#endif

  int lane = threadIdx.x & 31;
  int half = lane >> 4;
  int col  = lane & 15;
  int b0   = blockIdx.x * 16;

#if USE_TDM
  {
    unsigned soff = (unsigned)(unsigned long long)(void*)sStage;
    tdm_load_2d_f32(Wx, soff, 64, 192);
    __builtin_amdgcn_s_wait_tensorcnt(0);
    asm volatile("" ::: "memory");
    for (int f = 0; f < 24; ++f) {        // frag f = ntile*2 + kchunk
      int nt = f >> 1, kc = f & 1;
      store_frag(sFX, f, lane, load_b_lds_f32(sStage, 192, kc * 32, nt * 16, lane));
    }
    asm volatile("s_wait_dscnt 0x0" ::: "memory");  // staging reads done before reuse
    tdm_load_2d_f32(Wh, soff, 64, 192);
    __builtin_amdgcn_s_wait_tensorcnt(0);
    asm volatile("" ::: "memory");
    for (int f = 0; f < 24; ++f) {
      int nt = f >> 1, kc = f & 1;
      store_frag(sFH, f, lane, load_b_lds_f32(sStage, 192, kc * 32, nt * 16, lane));
    }
  }
#else
  for (int f = 0; f < 24; ++f) {
    int nt = f >> 1, kc = f & 1;
    store_frag(sFX, f, lane, load_b_global(Wx, 64, 192, kc * 32, nt * 16, lane));
    store_frag(sFH, f, lane, load_b_global(Wh, 64, 192, kc * 32, nt * 16, lane));
  }
#endif
  for (int i = lane; i < 16 * 64; i += 32) sH[i] = to_bf16(0.f);

  float biasv[12];
#pragma unroll
  for (int nt = 0; nt < 12; ++nt) biasv[nt] = bias[nt * 16 + col];
  int lenv[8];
#pragma unroll
  for (int r = 0; r < 8; ++r) lenv[r] = hist_len[b0 + r + half * 8];

  float hC[4][8];
#pragma unroll
  for (int jt = 0; jt < 4; ++jt)
#pragma unroll
    for (int r = 0; r < 8; ++r) hC[jt][r] = 0.f;

  __syncthreads();

  for (int t = 0; t < NT; ++t) {
    v16bf xA0, xA1, hA0, hA1;
    if (!AUGRU) {
      int xi = seq_idx[(long)(b0 + (lane & 15)) * NT + t];
      load_a_row64(xA0, xA1, emb + (long)xi * ND, lane);
    } else {
      load_a_row64(xA0, xA1, seq_in + ((long)(b0 + (lane & 15)) * NT + t) * ND, lane);
    }
    if (t + 1 < NT) {                       // prefetch next timestep's input row
      const float* nx = AUGRU
          ? seq_in + ((long)(b0 + (lane & 15)) * NT + t + 1) * ND
          : emb + (long)seq_idx[(long)(b0 + (lane & 15)) * NT + t + 1] * ND;
      __builtin_prefetch(nx, 0, 1);
    }
    load_a_lds_bf16_64(hA0, hA1, sH, lane);

    float att8[8];
#pragma unroll
    for (int r = 0; r < 8; ++r)
      att8[r] = AUGRU ? scores[(long)(b0 + r + half * 8) * NT + t] : 1.0f;

#pragma unroll
    for (int jt = 0; jt < 4; ++jt) {        // hidden-column tile 0..3
      v8f az, ar, ah, gz, gr, gh;
#pragma unroll
      for (int i = 0; i < 8; ++i) {
        az[i] = biasv[jt]; ar[i] = biasv[jt + 4]; ah[i] = biasv[jt + 8];
        gz[i] = 0.f; gr[i] = 0.f; gh[i] = 0.f;
      }
      az = wmma_bf16(xA0, load_frag(sFX, jt * 2 + 0, lane), az);
      az = wmma_bf16(xA1, load_frag(sFX, jt * 2 + 1, lane), az);
      ar = wmma_bf16(xA0, load_frag(sFX, (jt + 4) * 2 + 0, lane), ar);
      ar = wmma_bf16(xA1, load_frag(sFX, (jt + 4) * 2 + 1, lane), ar);
      ah = wmma_bf16(xA0, load_frag(sFX, (jt + 8) * 2 + 0, lane), ah);
      ah = wmma_bf16(xA1, load_frag(sFX, (jt + 8) * 2 + 1, lane), ah);
      gz = wmma_bf16(hA0, load_frag(sFH, jt * 2 + 0, lane), gz);
      gz = wmma_bf16(hA1, load_frag(sFH, jt * 2 + 1, lane), gz);
      gr = wmma_bf16(hA0, load_frag(sFH, (jt + 4) * 2 + 0, lane), gr);
      gr = wmma_bf16(hA1, load_frag(sFH, (jt + 4) * 2 + 1, lane), gr);
      gh = wmma_bf16(hA0, load_frag(sFH, (jt + 8) * 2 + 0, lane), gh);
      gh = wmma_bf16(hA1, load_frag(sFH, (jt + 8) * 2 + 1, lane), gh);

#pragma unroll
      for (int r = 0; r < 8; ++r) {
        float z    = sig_(az[r] + gz[r]) * att8[r];
        float rr   = sig_(ar[r] + gr[r]);
        float cand = tanhf(ah[r] + rr * gh[r]);
        float hp   = hC[jt][r];
        float hn   = (1.f - z) * hp + z * cand;
        hC[jt][r]  = (t < lenv[r]) ? hn : hp;
      }
    }
    __syncthreads();
#pragma unroll
    for (int jt = 0; jt < 4; ++jt)
#pragma unroll
      for (int r = 0; r < 8; ++r) {
        int row = r + half * 8;
        sH[row * 64 + jt * 16 + col] = to_bf16(hC[jt][r]);
        if (!AUGRU)
          seq_h_out[((long)(b0 + row) * NT + t) * ND + jt * 16 + col] = hC[jt][r];
      }
    __syncthreads();
  }

  if (AUGRU) {
#pragma unroll
    for (int jt = 0; jt < 4; ++jt)
#pragma unroll
      for (int r = 0; r < 8; ++r)
        final_out[(long)(b0 + r + half * 8) * ND + jt * 16 + col] = hC[jt][r];
  }
}

// ---------------------------------------------------------------- attention
__global__ __launch_bounds__(32) void attention_kernel(
    const float* __restrict__ seq_h, const int* __restrict__ seq_idx,
    const int* __restrict__ item_idx, const float* __restrict__ emb_item,
    const float* __restrict__ W1, const float* __restrict__ b1,
    const float* __restrict__ W2, const float* __restrict__ b2,
    const float* __restrict__ W3, const float* __restrict__ b3,
    float* __restrict__ scores)
{
  __shared__ __attribute__((aligned(32))) __bf16 sF1[32 * 32 * 16];
  __shared__ __attribute__((aligned(32))) __bf16 sF2[2 * 32 * 16];
  __shared__ __attribute__((aligned(32))) __bf16 sH1[16 * 64];
  __shared__ float sQ[64];
  __shared__ __attribute__((aligned(16))) float sK[16 * 64];
  __shared__ float sH2[16 * 16];
  __shared__ float sLog[112];
  __shared__ float sTmp[112];

  int lane = threadIdx.x & 31;
  int half = lane >> 4;
  int col  = lane & 15;
  int b    = blockIdx.x;

  for (int f = 0; f < 32; ++f) {           // frag f = kchunk*4 + ntile
    int kc = f >> 2, nt = f & 3;
    store_frag(sF1, f, lane, load_b_global(W1, 256, 64, kc * 32, nt * 16, lane));
  }
  store_frag(sF2, 0, lane, load_b_global(W2, 64, 16, 0, 0, lane));
  store_frag(sF2, 1, lane, load_b_global(W2, 64, 16, 32, 0, lane));
  {
    long q = (long)item_idx[b] * ND;
    for (int i = lane; i < 64; i += 32) sQ[i] = emb_item[q + i];
  }
  __syncthreads();

  for (int mt = 0; mt < 7; ++mt) {
    for (int i = lane; i < 16 * 64; i += 32) {
      int tt = mt * 16 + (i >> 6);
      sK[i] = (tt < NT) ? seq_h[((long)b * NT + tt) * ND + (i & 63)] : 0.f;
    }
    if (mt + 1 < 7) {
      int tn = (mt + 1) * 16 + (lane & 15);
      if (tn < NT) __builtin_prefetch(seq_h + ((long)b * NT + tn) * ND, 0, 1);
    }
    __syncthreads();

    v16bf aF[8];                           // info tile A-frags, K = 256
#pragma unroll
    for (int kc = 0; kc < 8; ++kc) {
#pragma unroll
      for (int j = 0; j < 8; ++j) {
        int k = kc * 32 + ((j & 3) << 1) + ((j >> 2) << 4) + (half << 3);
#pragma unroll
        for (int e = 0; e < 2; ++e) {
          int kk = k + e;
          int sec = kk >> 6, kd = kk & 63;
          float qv = sQ[kd], kv = sK[(lane & 15) * 64 + kd];
          float v = (sec == 0) ? qv : (sec == 1) ? kv : (sec == 2) ? (qv - kv) : (qv * kv);
          aF[kc][2 * j + e] = to_bf16(v);
        }
      }
    }

#pragma unroll
    for (int nt = 0; nt < 4; ++nt) {
      v8f acc; float bb = b1[nt * 16 + col];
#pragma unroll
      for (int i = 0; i < 8; ++i) acc[i] = bb;
#pragma unroll
      for (int kc = 0; kc < 8; ++kc)
        acc = wmma_bf16(aF[kc], load_frag(sF1, kc * 4 + nt, lane), acc);
#pragma unroll
      for (int r = 0; r < 8; ++r)
        sH1[(r + half * 8) * 64 + nt * 16 + col] = to_bf16(sig_(acc[r]));
    }
    __syncthreads();

    {
      v16bf hA0, hA1;
      load_a_lds_bf16_64(hA0, hA1, sH1, lane);
      v8f acc; float bb = b2[col];
#pragma unroll
      for (int i = 0; i < 8; ++i) acc[i] = bb;
      acc = wmma_bf16(hA0, load_frag(sF2, 0, lane), acc);
      acc = wmma_bf16(hA1, load_frag(sF2, 1, lane), acc);
#pragma unroll
      for (int r = 0; r < 8; ++r)
        sH2[(r + half * 8) * 16 + col] = sig_(acc[r]);
    }
    __syncthreads();

    if (lane < 16) {
      int tg = mt * 16 + lane;
      if (tg < NT) {
        float lg = b3[0];
#pragma unroll
        for (int c = 0; c < 16; ++c) lg += sH2[lane * 16 + c] * W3[c];
        bool msk = seq_idx[(long)b * NT + tg] != 0;
        sLog[tg] = msk ? lg : -1.0e9f;
      }
    }
    __syncthreads();
  }

  float mx = -3.4e38f;
  for (int t = lane; t < NT; t += 32) mx = fmaxf(mx, sLog[t]);
#pragma unroll
  for (int off = 16; off > 0; off >>= 1) mx = fmaxf(mx, __shfl_xor(mx, off, 32));
  float sm = 0.f;
  for (int t = lane; t < NT; t += 32) { float e = __expf(sLog[t] - mx); sTmp[t] = e; sm += e; }
#pragma unroll
  for (int off = 16; off > 0; off >>= 1) sm += __shfl_xor(sm, off, 32);
  __syncthreads();
  for (int t = lane; t < NT; t += 32) scores[(long)b * NT + t] = sTmp[t] / sm;
}

// ---------------------------------------------------------------- aux loss
__global__ void aux_kernel(const float* __restrict__ seq_h,
                           const int* __restrict__ seq_idx,
                           const int* __restrict__ neg_idx,
                           const float* __restrict__ emb_item,
                           const float* __restrict__ emb_neg,
                           float* __restrict__ sums)
{
  int total = NB * (NT - 1);
  float ls = 0.f, ms = 0.f;
  for (int i = blockIdx.x * blockDim.x + threadIdx.x; i < total; i += gridDim.x * blockDim.x) {
    int b = i / (NT - 1), tt = i % (NT - 1);
    int pidx = seq_idx[(long)b * NT + tt + 1];
    if (pidx != 0) {
      const float* h  = seq_h + ((long)b * NT + tt) * ND;
      const float* pe = emb_item + (long)pidx * ND;
      const float* ne = emb_neg + (long)neg_idx[(long)b * NT + tt + 1] * ND;
      float dp = 0.f, dn = 0.f;
      for (int k = 0; k < ND; ++k) { dp += h[k] * pe[k]; dn += h[k] * ne[k]; }
      ls += __logf(sig_(dp) + 1e-8f) + __logf(1.f - sig_(dn) + 1e-8f);
      ms += 1.f;
    }
  }
#pragma unroll
  for (int off = 16; off > 0; off >>= 1) {
    ls += __shfl_xor(ls, off, 32);
    ms += __shfl_xor(ms, off, 32);
  }
  if ((threadIdx.x & 31) == 0) {
    atomicAdd(&sums[0], ls);
    atomicAdd(&sums[1], ms);
  }
}

__global__ void zero_kernel(float* sums) { sums[0] = 0.f; sums[1] = 0.f; }
__global__ void aux_final_kernel(const float* sums, float* out) {
  out[NB] = -sums[0] / (sums[1] + 1e-8f);
}

// ---------------------------------------------------------------- FFN head
__global__ __launch_bounds__(32) void ffn_kernel(
    const float* __restrict__ finals, const int* __restrict__ item_idx,
    const float* __restrict__ emb_item, const float* __restrict__ dense,
    const int* __restrict__ sparse, const float* __restrict__ emb_other,
    const float* __restrict__ gamma, const float* __restrict__ beta,
    const float* __restrict__ W0, const float* __restrict__ fb0, const float* __restrict__ da0,
    const float* __restrict__ W1, const float* __restrict__ fb1, const float* __restrict__ da1,
    const float* __restrict__ W2, const float* __restrict__ fb2, const float* __restrict__ da2,
    const float* __restrict__ oW, const float* __restrict__ ob,
    float* __restrict__ out)
{
  __shared__ __attribute__((aligned(16))) float sA[16 * 256];
  __shared__ __attribute__((aligned(16))) float sB[16 * 256];
  const float INV = 0.99950037f;           // 1/sqrt(1 + 0.001)

  int lane = threadIdx.x & 31;
  int half = lane >> 4;
  int col  = lane & 15;
  int b0   = blockIdx.x * 16;

  for (int i = lane; i < 16 * 256; i += 32) { sA[i] = 0.f; sB[i] = 0.f; }
  __syncthreads();

  for (int i = lane; i < 16 * 200; i += 32) {
    int m = i / 200, c = i % 200;
    int gb = b0 + m;
    float v;
    if (c < 64)       v = finals[(long)gb * 64 + c];
    else if (c < 128) v = emb_item[(long)item_idx[gb] * 64 + (c - 64)];
    else if (c < 136) v = dense[(long)gb * 8 + (c - 128)];
    else if (c < 168) v = emb_other[(long)sparse[gb * 2 + 0] * 32 + (c - 136)];
    else              v = emb_other[(long)10000 * 32 + (long)sparse[gb * 2 + 1] * 32 + (c - 168)];
    sA[m * 256 + c] = gamma[c] * v * INV + beta[c];
  }
  __syncthreads();

  const float* Ws[3]  = {W0, W1, W2};
  const float* bs[3]  = {fb0, fb1, fb2};
  const float* as_[3] = {da0, da1, da2};
  const int Ks[3] = {200, 256, 128};
  const int Ns[3] = {256, 128, 64};
  float* cur = sA; float* nxt = sB;

  for (int L = 0; L < 3; ++L) {
    int K = Ks[L], N = Ns[L], KC = (K + 31) >> 5;
    for (int nt = 0; nt < (N >> 4); ++nt) {
      int n = nt * 16 + col;
      v8f acc; float bb = bs[L][n];
#pragma unroll
      for (int i = 0; i < 8; ++i) acc[i] = bb;
      for (int kc = 0; kc < KC; ++kc) {
        v16bf a = load_a_lds_f32(cur, 256, kc * 32, lane);
        v16bf w = load_b_global(Ws[L], K, N, kc * 32, nt * 16, lane);
        acc = wmma_bf16(a, w, acc);
      }
      float al = as_[L][n];
#pragma unroll
      for (int r = 0; r < 8; ++r) {        // dice activation
        float y = acc[r];
        float p = sig_(y * INV);
        nxt[(r + half * 8) * 256 + n] = p * y + (1.f - p) * al * y;
      }
    }
    __syncthreads();
    float* tmp = cur; cur = nxt; nxt = tmp;
    __syncthreads();
  }

  if (lane < 16) {
    float o = ob[0];
    for (int c = 0; c < 64; ++c) o += cur[lane * 256 + c] * oW[c];
    out[b0 + lane] = sig_(o);
  }
}

// ---------------------------------------------------------------- launcher
extern "C" void kernel_launch(void* const* d_in, const int* in_sizes, int n_in,
                              void* d_out, int out_size, void* d_ws, size_t ws_size,
                              hipStream_t stream) {
  (void)in_sizes; (void)n_in; (void)out_size; (void)ws_size;
  const float* dense    = (const float*)d_in[0];
  const int*   sparse   = (const int*)d_in[1];
  const int*   seq_idx  = (const int*)d_in[2];
  const int*   neg_idx  = (const int*)d_in[3];
  const int*   item_idx = (const int*)d_in[4];
  const int*   hist_len = (const int*)d_in[5];
  const float* emb_item = (const float*)d_in[6];
  const float* emb_neg  = (const float*)d_in[7];
  const float* emb_oth  = (const float*)d_in[8];
  const float* g1Wx = (const float*)d_in[9];
  const float* g1Wh = (const float*)d_in[10];
  const float* g1b  = (const float*)d_in[11];
  const float* aW1 = (const float*)d_in[12]; const float* ab1 = (const float*)d_in[13];
  const float* aW2 = (const float*)d_in[14]; const float* ab2 = (const float*)d_in[15];
  const float* aW3 = (const float*)d_in[16]; const float* ab3 = (const float*)d_in[17];
  const float* g2Wx = (const float*)d_in[18];
  const float* g2Wh = (const float*)d_in[19];
  const float* g2b  = (const float*)d_in[20];
  const float* gamma = (const float*)d_in[21]; const float* beta = (const float*)d_in[22];
  const float* W0 = (const float*)d_in[23]; const float* fb0 = (const float*)d_in[24];
  const float* da0 = (const float*)d_in[25];
  const float* W1 = (const float*)d_in[26]; const float* fb1 = (const float*)d_in[27];
  const float* da1 = (const float*)d_in[28];
  const float* W2 = (const float*)d_in[29]; const float* fb2 = (const float*)d_in[30];
  const float* da2 = (const float*)d_in[31];
  const float* oW = (const float*)d_in[32]; const float* ob = (const float*)d_in[33];

  float* ws     = (float*)d_ws;
  float* seq_h  = ws;                                 // B*T*64
  float* scores = seq_h + (size_t)NB * NT * ND;       // B*T
  float* finals = scores + (size_t)NB * NT;           // B*64
  float* sums   = finals + (size_t)NB * ND;           // 2
  float* out    = (float*)d_out;                      // 2048 probs + 1 aux

  zero_kernel<<<1, 1, 0, stream>>>(sums);
  gru_kernel<false><<<NB / 16, 32, 0, stream>>>(
      seq_idx, emb_item, nullptr, nullptr, hist_len, g1Wx, g1Wh, g1b, seq_h, nullptr);
  aux_kernel<<<512, 256, 0, stream>>>(seq_h, seq_idx, neg_idx, emb_item, emb_neg, sums);
  attention_kernel<<<NB, 32, 0, stream>>>(
      seq_h, seq_idx, item_idx, emb_item, aW1, ab1, aW2, ab2, aW3, ab3, scores);
  gru_kernel<true><<<NB / 16, 32, 0, stream>>>(
      nullptr, nullptr, seq_h, scores, hist_len, g2Wx, g2Wh, g2b, nullptr, finals);
  ffn_kernel<<<NB / 16, 32, 0, stream>>>(
      finals, item_idx, emb_item, dense, sparse, emb_oth, gamma, beta,
      W0, fb0, da0, W1, fb1, da1, W2, fb2, da2, oW, ob, out);
  aux_final_kernel<<<1, 1, 0, stream>>>(sums, out);
}
